// HeuristicGNN_7473243095608
// MI455X (gfx1250) — matrix operand order repaired and checked by reading, not compiled, not run on hardware
//
#include <hip/hip_runtime.h>
#include <hip/hip_bf16.h>

// CDNA5 / gfx1250, wave32. Bandwidth-bound kernel (512 MiB streamed, ~22us floor
// at 23.3 TB/s). WMMA bf16 does the weighted row-reduction with the dense mask
// as the fully-utilized A operand; per-i edge weights are single-nonzero-column
// B operands accumulated into a 16(batch) x 16(i) f32 tile.

typedef __attribute__((ext_vector_type(16))) __bf16 v16bf;
typedef __attribute__((ext_vector_type(8)))  float  v8f;
typedef __attribute__((ext_vector_type(4)))  float  v4f;

#define KOBJ 32   // K objects

// ---------------------------------------------------------------------------
// Tiny setup kernel: edges[i][j] = relu(E_w[0,i] + E_w[0,32+j] + E_b)
// (one-hot features make the edge MLP batch-independent)
// ---------------------------------------------------------------------------
__global__ void edges_kernel(const float* __restrict__ E_w,
                             const float* __restrict__ E_b,
                             float* __restrict__ edges) {
  int t = threadIdx.x;            // 1024 threads, one per (i,j)
  int i = t >> 5, j = t & 31;
  float v = E_w[i] + E_w[KOBJ + j] + E_b[0];
  edges[t] = fmaxf(v, 0.0f);
}

// ---------------------------------------------------------------------------
// Load one lane's half of a 16x32 A tile row (f32 in memory) and pack to bf16
// in the documented 16-bit A layout:
//   lane<16 : j in {0..7} U {16..23}   lane>=16 : j in {8..15} U {24..31}
// p points at mask[n, i, jb] with jb = 0 or 8; chunks at +0..7 and +16..23.
// ---------------------------------------------------------------------------
static __device__ __forceinline__ v16bf load_cvt_a(const float* __restrict__ p) {
  v4f x0 = *(const v4f*)(p + 0);
  v4f x1 = *(const v4f*)(p + 4);
  v4f x2 = *(const v4f*)(p + 16);
  v4f x3 = *(const v4f*)(p + 20);
  v16bf a;
#pragma unroll
  for (int t = 0; t < 4; ++t) {
    a[t]      = (__bf16)x0[t];
    a[4 + t]  = (__bf16)x1[t];
    a[8 + t]  = (__bf16)x2[t];
    a[12 + t] = (__bf16)x3[t];
  }
  return a;
}

// ---------------------------------------------------------------------------
// Main kernel: one wave32 per 16 batch elements.
// ---------------------------------------------------------------------------
__launch_bounds__(256)
__global__ void gnn_kernel(const float* __restrict__ sm,   // state mask (N,32,32)
                           const float* __restrict__ gm,   // goal mask
                           const float* __restrict__ edges,// (32,32) f32
                           const float* __restrict__ N_w,  // (1,33)
                           const float* __restrict__ N_b,  // (1,)
                           const float* __restrict__ O1_w, // (1,2)
                           const float* __restrict__ O1_b,
                           const float* __restrict__ O2_w,
                           const float* __restrict__ O2_b,
                           float* __restrict__ out) {
  const int lane = threadIdx.x & 31;
  const int wave = threadIdx.x >> 5;
  const int tile = blockIdx.x * 8 + wave;   // 8 waves per block
  const int n0   = tile * 16;               // 16 batch rows per wave
  const int c    = lane & 15;               // B/D column index (i mod 16)
  const int hi   = lane >> 4;               // half-wave id

  // --- Build per-lane B operands. Column c of B holds edges[i,:]; per lane
  // only i = c and i = c+16 are ever nonzero, so 2 vectors cover all 32 B_i.
  // B striping: lanes 0-15 hold K=0..7,16..23, lanes 16-31 hold K=8..15,24..31.
  const int kbase = hi ? 8 : 0;
  v16bf blo, bhi;
  {
    const float* e0 = edges + c * KOBJ + kbase;          // edges[c, kbase..]
    const float* e1 = edges + (16 + c) * KOBJ + kbase;   // edges[16+c, kbase..]
#pragma unroll
    for (int t = 0; t < 16; ++t) {
      blo[t] = (__bf16)e0[t < 8 ? t : t + 8];
      bhi[t] = (__bf16)e1[t < 8 ? t : t + 8];
    }
  }
  const v16bf zerov = {};

  // --- Per-lane mask row base: A row m = lane%16 -> batch n0 + c.
  const size_t rowOff = (size_t)(n0 + c) * (KOBJ * KOBJ);
  const float* srow = sm + rowOff;
  const float* grow = gm + rowOff;
  const int jb = hi ? 8 : 0;

  v8f d0s = {}, d1s = {}, d0g = {}, d1g = {};

  // ---- state mask: he_state tile (prefetch goal rows to warm L2) ----
#pragma unroll 4
  for (int i = 0; i < 16; ++i) {
    v16bf a = load_cvt_a(srow + i * KOBJ + jb);
    __builtin_prefetch(grow + i * KOBJ + jb, 0, 0);
    v16bf b = (c == i) ? blo : zerov;
    d0s = __builtin_amdgcn_wmma_f32_16x16x32_bf16(false, a, false, b,
                                                  (short)0, d0s, false, false);
  }
#pragma unroll 4
  for (int i = 16; i < 32; ++i) {
    v16bf a = load_cvt_a(srow + i * KOBJ + jb);
    __builtin_prefetch(grow + i * KOBJ + jb, 0, 0);
    v16bf b = (c == (i - 16)) ? bhi : zerov;
    d1s = __builtin_amdgcn_wmma_f32_16x16x32_bf16(false, a, false, b,
                                                  (short)0, d1s, false, false);
  }
  // ---- goal mask: he_goal tile ----
#pragma unroll 4
  for (int i = 0; i < 16; ++i) {
    v16bf a = load_cvt_a(grow + i * KOBJ + jb);
    v16bf b = (c == i) ? blo : zerov;
    d0g = __builtin_amdgcn_wmma_f32_16x16x32_bf16(false, a, false, b,
                                                  (short)0, d0g, false, false);
  }
#pragma unroll 4
  for (int i = 16; i < 32; ++i) {
    v16bf a = load_cvt_a(grow + i * KOBJ + jb);
    v16bf b = (c == (i - 16)) ? bhi : zerov;
    d1g = __builtin_amdgcn_wmma_f32_16x16x32_bf16(false, a, false, b,
                                                  (short)0, d1g, false, false);
  }

  // --- Epilogue: node MLP + mean over i + output MLP.
  // D layout: lane c holds he[n0+r, ibase+c] in vgpr r (r=0..7) for half hi=0,
  // he[n0+8+r, ...] for hi=1. Sum over i == shuffle-reduce over 16 lanes.
  const float nw32 = N_w[KOBJ];
  const float nc0  = N_w[c]      + N_b[0];   // bias+onehot weight for i=c
  const float nc1  = N_w[16 + c] + N_b[0];   // for i=16+c
  const float o1w0 = O1_w[0], o1w1 = O1_w[1], o1b = O1_b[0];
  const float o2w  = O2_w[0], o2b  = O2_b[0];

#pragma unroll
  for (int r = 0; r < 8; ++r) {
    float hs = fmaxf(fmaf(d0s[r], nw32, nc0), 0.0f) +
               fmaxf(fmaf(d1s[r], nw32, nc1), 0.0f);
    float hg = fmaxf(fmaf(d0g[r], nw32, nc0), 0.0f) +
               fmaxf(fmaf(d1g[r], nw32, nc1), 0.0f);
#pragma unroll
    for (int m = 8; m >= 1; m >>= 1) {        // reduce within each 16-lane half
      hs += __shfl_xor(hs, m, 32);
      hg += __shfl_xor(hg, m, 32);
    }
    hs *= (1.0f / 32.0f);
    hg *= (1.0f / 32.0f);
    float pred = fmaxf(fmaf(hs, o1w0, fmaf(hg, o1w1, o1b)), 0.0f) * o2w + o2b;
    if (c == 0) out[n0 + hi * 8 + r] = pred;  // lanes 0 and 16 write
  }
}

// ---------------------------------------------------------------------------
extern "C" void kernel_launch(void* const* d_in, const int* in_sizes, int n_in,
                              void* d_out, int out_size, void* d_ws, size_t ws_size,
                              hipStream_t stream) {
  const float* sm   = (const float*)d_in[0];
  const float* gm   = (const float*)d_in[1];
  const float* E_w  = (const float*)d_in[2];
  const float* E_b  = (const float*)d_in[3];
  const float* N_w  = (const float*)d_in[4];
  const float* N_b  = (const float*)d_in[5];
  const float* O1_w = (const float*)d_in[6];
  const float* O1_b = (const float*)d_in[7];
  const float* O2_w = (const float*)d_in[8];
  const float* O2_b = (const float*)d_in[9];
  float* out   = (float*)d_out;
  float* edges = (float*)d_ws;    // 32*32 f32 = 4 KiB scratch

  const int N = in_sizes[0] / (KOBJ * KOBJ);   // 65536

  edges_kernel<<<1, KOBJ * KOBJ, 0, stream>>>(E_w, E_b, edges);

  // 8 waves/block * 16 batch/wave = 128 batch per block
  const int nblocks = N / 128;                 // 512, exact for N=65536
  gnn_kernel<<<nblocks, 256, 0, stream>>>(sm, gm, edges, N_w, N_b,
                                          O1_w, O1_b, O2_w, O2_b, out);
}
